// MultiHeadAttention_88235808129669
// MI455X (gfx1250) — matrix operand order repaired and checked by reading, not compiled, not run on hardware
//
#include <hip/hip_runtime.h>
#include <stdint.h>

typedef __bf16 bf16_t;
typedef __attribute__((ext_vector_type(16))) __bf16 v16bf;
typedef __attribute__((ext_vector_type(8)))  float   v8f;
typedef int v4i_t __attribute__((vector_size(16)));   // matches builtin pointee
typedef __attribute__((address_space(1))) v4i_t* gv4i_p;   // non-const, per diagnostic
typedef __attribute__((address_space(3))) v4i_t* sv4i_p;

#define DEV __device__ __forceinline__

constexpr int kB = 2, kS = 2048, kD = 256, kH = 8;
constexpr int kHD = kH * kD;              // 2048
constexpr float kQScale = 0.0625f;        // 1/sqrt(256), folded into Wq/bq

#if __has_builtin(__builtin_amdgcn_global_load_async_to_lds_b128) && \
    __has_builtin(__builtin_amdgcn_s_wait_asynccnt)
#define USE_ASYNC_LDS 1
#endif

// ---- helpers ---------------------------------------------------------------

DEV bf16_t f2bf(float f) {
  union { float f; uint32_t u; } a; a.f = f;
  uint32_t r = a.u + 0x7FFFu + ((a.u >> 16) & 1u);   // round-to-nearest-even
  union { uint16_t s; bf16_t b; } o; o.s = (uint16_t)(r >> 16);
  return o.b;
}

// A fragment (16x32, row-major source): lane&15 = row M,
// lanes 0-15 hold K=0..7,16..23 ; lanes 16-31 hold K=8..15,24..31
DEV v16bf load_a(const bf16_t* p, int ld) {
  int lane = threadIdx.x & 31;
  const bf16_t* q = p + (size_t)(lane & 15) * ld + ((lane >> 4) << 3);
  union { uint4 u[2]; v16bf v; } f;
  f.u[0] = *reinterpret_cast<const uint4*>(q);
  f.u[1] = *reinterpret_cast<const uint4*>(q + 16);
  return f.v;
}

// B fragment (32x16), loaded from B^T stored row-major [N][K]:
// lane&15 = col N, lanes 0-15 hold K=0..7,16..23 ; lanes 16-31 hold K=8..15,24..31
DEV v16bf load_b(const bf16_t* p, int ld) {
  int lane = threadIdx.x & 31;
  const bf16_t* q = p + (size_t)(lane & 15) * ld + ((lane >> 4) << 3);
  union { uint4 u[2]; v16bf v; } f;
  f.u[0] = *reinterpret_cast<const uint4*>(q);
  f.u[1] = *reinterpret_cast<const uint4*>(q + 16);
  return f.v;
}

DEV v8f wmma_bf16(v16bf a, v16bf b, v8f c) {
  return __builtin_amdgcn_wmma_f32_16x16x32_bf16(false, a, false, b,
                                                 (short)0, c, false, false);
}

// 16-byte global -> LDS copy (async-to-LDS when available)
DEV void cp16(bf16_t* lds, const bf16_t* g) {
#ifdef USE_ASYNC_LDS
  __builtin_amdgcn_global_load_async_to_lds_b128(
      (gv4i_p)(uintptr_t)g,
      (sv4i_p)(uint32_t)(uintptr_t)lds,   // low 32 bits = LDS offset (ISA aperture rule)
      0, 0);
#else
  *reinterpret_cast<uint4*>(lds) = *reinterpret_cast<const uint4*>(g);
#endif
}

DEV void cp16_flush() {
#ifdef USE_ASYNC_LDS
  __builtin_amdgcn_s_wait_asynccnt(0);
#endif
}

// ---- prep kernels ----------------------------------------------------------

__global__ void k_cvt(const float* __restrict__ src, bf16_t* __restrict__ dst, int n) {
  int i = blockIdx.x * blockDim.x + threadIdx.x;
  if (i < n) dst[i] = f2bf(src[i]);
}

// W [H][Din][Dout] -> WT [H][Dout][Din] * scale (bf16)
__global__ void k_tw(const float* __restrict__ src, bf16_t* __restrict__ dst,
                     float scale, int n) {
  int i = blockIdx.x * blockDim.x + threadIdx.x;
  if (i >= n) return;
  int d = i & (kD - 1);
  int e = (i >> 8) & (kD - 1);
  int h = i >> 16;
  dst[i] = f2bf(src[(h * kD + d) * kD + e] * scale);
}

// Wo [HD][D] -> WoT [D][HD] (bf16)
__global__ void k_two(const float* __restrict__ src, bf16_t* __restrict__ dst, int n) {
  int i = blockIdx.x * blockDim.x + threadIdx.x;
  if (i >= n) return;
  int f = i & (kHD - 1);
  int e = i >> 11;
  dst[i] = f2bf(src[f * kD + e]);
}

// ---- QKV projection: one wave = 16x64 tile of one of Q/K/V ----------------

__global__ __launch_bounds__(256) void k_qkv(
    const bf16_t* __restrict__ xb, const bf16_t* __restrict__ wT,
    const float* __restrict__ bq, const float* __restrict__ bk,
    const float* __restrict__ bv,
    bf16_t* __restrict__ Qb, bf16_t* __restrict__ Kb, bf16_t* __restrict__ Vt) {
  int wid = threadIdx.x >> 5, lane = threadIdx.x & 31;
  int gw = blockIdx.x * 8 + wid;          // 24576 waves total
  int n0 = (gw & 3) * 64;
  int qt = (gw >> 2) & 127;
  int bh = (gw >> 9) & 15;                // b*8 + h
  int w  = gw >> 13;                      // 0=Q 1=K 2=V
  int b = bh >> 3, h = bh & 7;
  int n = lane & 15, hi = lane >> 4;

  const bf16_t* Abase = xb + (size_t)(b * kS + qt * 16) * kD;
  const bf16_t* Bbase = wT + (size_t)((w * kH + h) * kD + n0) * kD;
  const float* bias = (w == 0) ? bq : (w == 1) ? bk : bv;
  float bscale = (w == 0) ? kQScale : 1.0f;

  v8f z = {};
  v8f c[4]; for (int j = 0; j < 4; ++j) c[j] = z;
  for (int kb = 0; kb < kD; kb += 32) {
    v16bf a = load_a(Abase + kb, kD);
    #pragma unroll
    for (int j = 0; j < 4; ++j) {
      v16bf bb = load_b(Bbase + (size_t)(16 * j) * kD + kb, kD);
      c[j] = wmma_bf16(a, bb, c[j]);
    }
  }
  if (w < 2) {                            // Q, K: row-major [b*H+h][S][D]
    bf16_t* dst = ((w == 0) ? Qb : Kb) + ((size_t)bh * kS + qt * 16) * kD;
    #pragma unroll
    for (int j = 0; j < 4; ++j) {
      float bs = bias[h * kD + n0 + 16 * j + n] * bscale;
      #pragma unroll
      for (int i = 0; i < 8; ++i)
        dst[(size_t)(i + 8 * hi) * kD + n0 + 16 * j + n] = f2bf(c[j][i] + bs);
    }
  } else {                                // V transposed: [b*H+h][D][S]
    #pragma unroll
    for (int j = 0; j < 4; ++j) {
      int e = n0 + 16 * j + n;
      float bs = bias[h * kD + e];
      union { bf16_t hv[8]; uint4 u; } pk;
      #pragma unroll
      for (int i = 0; i < 8; ++i) pk.hv[i] = f2bf(c[j][i] + bs);
      *reinterpret_cast<uint4*>(Vt + ((size_t)bh * kD + e) * kS + qt * 16 + 8 * hi) = pk.u;
    }
  }
}

// ---- flash attention: block = 4 waves sharing one (b,h); K/V staged in LDS -

__global__ __launch_bounds__(128) void k_attn(
    const bf16_t* __restrict__ Qb, const bf16_t* __restrict__ Kb,
    const bf16_t* __restrict__ Vt, bf16_t* __restrict__ concat) {
  // padded to dodge bank conflicts; all row strides stay 16B multiples
  __shared__ bf16_t Ksh[32][264];         // K rows t0..t0+31, ld=264 (528B)
  __shared__ bf16_t Vsh[256][40];         // V^T rows e, 32 keys used, ld=40 (80B)
  __shared__ bf16_t Pt[4][32][32];        // per-wave P^T staging [key][query]

  int wid = threadIdx.x >> 5, lane = threadIdx.x & 31;
  int tid = threadIdx.x;
  int bh = blockIdx.x >> 5;               // b*8 + h
  int qblk = blockIdx.x & 31;             // 32 query blocks of 64
  int q0 = (qblk * 4 + wid) * 16;         // this wave's 16 queries
  int b = bh >> 3, h = bh & 7;
  int n = lane & 15, hi = lane >> 4;

  const bf16_t* Qbase = Qb + ((size_t)bh * kS + q0) * kD;
  const bf16_t* Kbase = Kb + (size_t)bh * kS * kD;
  const bf16_t* Vbase = Vt + (size_t)bh * kD * kS;

  v16bf qf[8];
  #pragma unroll
  for (int kk = 0; kk < 8; ++kk) qf[kk] = load_a(Qbase + kk * 32, kD);

  v8f z = {};
  v8f o[16];
  #pragma unroll
  for (int j = 0; j < 16; ++j) o[j] = z;
  float m_i[8], l_i[8];
  #pragma unroll
  for (int i = 0; i < 8; ++i) { m_i[i] = -1e30f; l_i[i] = 0.0f; }

  for (int t0 = 0; t0 < kS; t0 += 32) {
    __syncthreads();                      // prior tile fully consumed
    // stage K (32x256) and V^T (256x32): 1024 16B chunks each, 8 per thread
    const bf16_t* gK = Kbase + (size_t)t0 * kD;
    #pragma unroll
    for (int r = 0; r < 8; ++r) {
      int c = tid + r * 128;
      int krow = c >> 5, kcol = (c & 31) * 8;
      cp16(&Ksh[krow][kcol], gK + (size_t)krow * kD + kcol);
      int e = c >> 2, vcol = (c & 3) * 8;
      cp16(&Vsh[e][vcol], Vbase + (size_t)e * kS + t0 + vcol);
    }
    cp16_flush();
    __syncthreads();                      // staged tile visible to all waves

    // scores for 32 keys (Q already carries 1/sqrt(D))
    v8f s0 = z, s1 = z;
    #pragma unroll
    for (int kk = 0; kk < 8; ++kk) {
      v16bf b0 = load_b(&Ksh[0][kk * 32], 264);
      v16bf b1 = load_b(&Ksh[16][kk * 32], 264);
      s0 = wmma_bf16(qf[kk], b0, s0);
      s1 = wmma_bf16(qf[kk], b1, s1);
    }
    // online softmax; row r = i + 8*hi lives in one 16-lane half
    float alpha[8];
    #pragma unroll
    for (int i = 0; i < 8; ++i) {
      float mx = fmaxf(s0[i], s1[i]);
      mx = fmaxf(mx, __shfl_xor(mx, 1, 32));
      mx = fmaxf(mx, __shfl_xor(mx, 2, 32));
      mx = fmaxf(mx, __shfl_xor(mx, 4, 32));
      mx = fmaxf(mx, __shfl_xor(mx, 8, 32));
      float mn = fmaxf(m_i[i], mx);
      alpha[i] = __expf(m_i[i] - mn);
      m_i[i] = mn;
      float p0 = __expf(s0[i] - mn);
      float p1 = __expf(s1[i] - mn);
      s0[i] = p0; s1[i] = p1;
      float r = p0 + p1;
      r += __shfl_xor(r, 1, 32);
      r += __shfl_xor(r, 2, 32);
      r += __shfl_xor(r, 4, 32);
      r += __shfl_xor(r, 8, 32);
      l_i[i] = l_i[i] * alpha[i] + r;
    }
    #pragma unroll
    for (int j = 0; j < 16; ++j)
      #pragma unroll
      for (int i = 0; i < 8; ++i) o[j][i] *= alpha[i];

    // C-layout -> A-fragment layout via LDS (store P^T packed, reload)
    union { bf16_t hv[8]; uint4 u; } pk;
    #pragma unroll
    for (int i = 0; i < 8; ++i) pk.hv[i] = f2bf(s0[i]);
    *reinterpret_cast<uint4*>(&Pt[wid][n][8 * hi]) = pk.u;
    #pragma unroll
    for (int i = 0; i < 8; ++i) pk.hv[i] = f2bf(s1[i]);
    *reinterpret_cast<uint4*>(&Pt[wid][n + 16][8 * hi]) = pk.u;
    __syncthreads();                      // P^T visible across lanes

    v16bf pf;
    int kb = 8 * hi;
    #pragma unroll
    for (int i = 0; i < 8; ++i) {
      pf[i]     = Pt[wid][kb + i][n];
      pf[i + 8] = Pt[wid][kb + 16 + i][n];
    }

    // O += P (16x32) @ V (32 keys x 256) from LDS-resident V^T
    #pragma unroll
    for (int j = 0; j < 16; ++j) {
      v16bf vb = load_b(&Vsh[16 * j][0], 40);
      o[j] = wmma_bf16(pf, vb, o[j]);
    }
  }

  float inv[8];
  #pragma unroll
  for (int i = 0; i < 8; ++i) inv[i] = 1.0f / l_i[i];
  bf16_t* dst = concat + (size_t)(b * kS + q0) * kHD + h * kD;
  #pragma unroll
  for (int j = 0; j < 16; ++j)
    #pragma unroll
    for (int i = 0; i < 8; ++i)
      dst[(size_t)(i + 8 * hi) * kHD + 16 * j + n] = f2bf(o[j][i] * inv[i]);
}

// ---- output projection: concat [4096,2048] @ Wo [2048,256] + bo -----------

__global__ __launch_bounds__(256) void k_out(
    const bf16_t* __restrict__ concat, const bf16_t* __restrict__ woT,
    const float* __restrict__ bo, float* __restrict__ out) {
  int wid = threadIdx.x >> 5, lane = threadIdx.x & 31;
  int gw = blockIdx.x * 8 + wid;          // 1024 waves
  int n0 = (gw & 3) * 64;
  int mt = gw >> 2;
  int n = lane & 15, hi = lane >> 4;
  const bf16_t* Abase = concat + (size_t)mt * 16 * kHD;

  v8f z = {};
  v8f c[4]; for (int j = 0; j < 4; ++j) c[j] = z;
  for (int kb = 0; kb < kHD; kb += 32) {
    v16bf a = load_a(Abase + kb, kHD);
    #pragma unroll
    for (int j = 0; j < 4; ++j) {
      v16bf bb = load_b(woT + (size_t)(n0 + 16 * j) * kHD + kb, kHD);
      c[j] = wmma_bf16(a, bb, c[j]);
    }
  }
  #pragma unroll
  for (int j = 0; j < 4; ++j) {
    float bs = bo[n0 + 16 * j + n];
    #pragma unroll
    for (int i = 0; i < 8; ++i)
      out[(size_t)(mt * 16 + i + 8 * hi) * kD + n0 + 16 * j + n] = c[j][i] + bs;
  }
}

// ---- host ------------------------------------------------------------------

extern "C" void kernel_launch(void* const* d_in, const int* in_sizes, int n_in,
                              void* d_out, int out_size, void* d_ws, size_t ws_size,
                              hipStream_t stream) {
  const float* x  = (const float*)d_in[0];
  const float* Wq = (const float*)d_in[1];
  const float* bq = (const float*)d_in[2];
  const float* Wk = (const float*)d_in[3];
  const float* bk = (const float*)d_in[4];
  const float* Wv = (const float*)d_in[5];
  const float* bv = (const float*)d_in[6];
  const float* Wo = (const float*)d_in[7];
  const float* bo = (const float*)d_in[8];
  float* out = (float*)d_out;

  size_t off = 0;
  auto carve = [&](size_t elems) -> bf16_t* {
    bf16_t* p = (bf16_t*)((char*)d_ws + off);
    off += (elems * sizeof(bf16_t) + 255) & ~(size_t)255;
    return p;
  };
  bf16_t* xb  = carve((size_t)kB * kS * kD);        // x in bf16
  bf16_t* wT  = carve((size_t)3 * kH * kD * kD);    // WqT|WkT|WvT
  bf16_t* woT = carve((size_t)kD * kHD);            // Wo^T
  bf16_t* Qb  = carve((size_t)kB * kH * kS * kD);
  bf16_t* Kb  = carve((size_t)kB * kH * kS * kD);
  bf16_t* Vt  = carve((size_t)kB * kH * kD * kS);   // V transposed
  bf16_t* cc  = carve((size_t)kB * kS * kHD);       // concat heads (bf16)

  int nx = kB * kS * kD;
  k_cvt<<<(nx + 255) / 256, 256, 0, stream>>>(x, xb, nx);
  int nw = kH * kD * kD;
  k_tw<<<(nw + 255) / 256, 256, 0, stream>>>(Wq, wT + 0 * (size_t)nw, kQScale, nw);
  k_tw<<<(nw + 255) / 256, 256, 0, stream>>>(Wk, wT + 1 * (size_t)nw, 1.0f, nw);
  k_tw<<<(nw + 255) / 256, 256, 0, stream>>>(Wv, wT + 2 * (size_t)nw, 1.0f, nw);
  int nwo = kD * kHD;
  k_two<<<(nwo + 255) / 256, 256, 0, stream>>>(Wo, woT, nwo);

  k_qkv<<<3072, 256, 0, stream>>>(xb, wT, bq, bk, bv, Qb, Kb, Vt);
  k_attn<<<512, 128, 0, stream>>>(Qb, Kb, Vt, cc);
  k_out<<<128, 256, 0, stream>>>(cc, woT, bo, out);
}
